// MultiHeadSelfAttentionL2_74835510166180
// MI455X (gfx1250) — matrix operand order repaired and checked by reading, not compiled, not run on hardware
//
#include <hip/hip_runtime.h>
#include <stdint.h>

// MI455X / gfx1250: wave32, bf16 WMMA 16x16x32, fp32 accumulate.

typedef __attribute__((ext_vector_type(16))) __bf16 v16bf;
typedef __attribute__((ext_vector_type(8)))  float  v8f;

#define B_   8
#define S_   1024
#define D_   768
#define H_   12
#define F_   64
#define OUT_ 768
#define MTOT (B_*S_)

__device__ __forceinline__ uint16_t f2bf(float x) {
  uint32_t u = __float_as_uint(x);
  uint32_t r = u + 0x7FFFu + ((u >> 16) & 1u);   // round-to-nearest-even
  return (uint16_t)(r >> 16);
}
__device__ __forceinline__ float bf2f(uint16_t b) {
  return __uint_as_float(((uint32_t)b) << 16);
}

union AB { v16bf v; uint32_t u[8]; };

// ---------------- prep kernels ----------------
__global__ void k_f32_to_bf16(const float* __restrict__ src, uint16_t* __restrict__ dst, int n) {
  int i = blockIdx.x * blockDim.x + threadIdx.x;
  int stride = gridDim.x * blockDim.x;
  for (; i < n; i += stride) dst[i] = f2bf(src[i]);
}

// w: [H][D][F]  ->  dstT: [(H*F)][D]   (GEMM column contiguous along D)
__global__ void k_wT_bf16(const float* __restrict__ w, uint16_t* __restrict__ dstT) {
  int i = blockIdx.x * blockDim.x + threadIdx.x;
  const int n = H_ * D_ * F_;
  int stride = gridDim.x * blockDim.x;
  for (; i < n; i += stride) {
    int h   = i / (D_ * F_);
    int rem = i % (D_ * F_);
    int d   = rem / F_;
    int f   = rem % F_;
    dstT[(size_t)(h * F_ + f) * D_ + d] = f2bf(w[i]);
  }
}

// ---------------- fused QKV projection GEMM ----------------
// Y[8192 x 2304] = xb * Wall ; column c = which*768 + h*64 + f.
// One wave computes a 32x64 tile (two M blocks share every B fetch).
__global__ void k_qkv(const uint16_t* __restrict__ xb, const uint16_t* __restrict__ wT,
                      uint16_t* __restrict__ qb, uint16_t* __restrict__ kb,
                      uint16_t* __restrict__ vT) {
  const int lane = threadIdx.x;
  const int col  = lane & 15;
  const int hl   = lane >> 4;
  const int m0   = blockIdx.x * 32;
  const int gy   = blockIdx.y;          // 0..35  == (which, head)
  const int which = gy / H_;
  const int h     = gy % H_;
  const size_t c0 = (size_t)gy * 64;

  v8f acc[2][4] = {};
  for (int kc = 0; kc < D_; kc += 32) {
    AB a[2];
#pragma unroll
    for (int mb = 0; mb < 2; ++mb)
#pragma unroll
      for (int v = 0; v < 8; ++v) {
        int kk = kc + hl * 16 + v * 2;
        a[mb].u[v] = *(const uint32_t*)(xb + (size_t)(m0 + mb * 16 + col) * D_ + kk);
      }
#pragma unroll
    for (int nt = 0; nt < 4; ++nt) {
      size_t c = c0 + nt * 16 + col;
      int kbase = kc + hl * 16;
      AB b;
#pragma unroll
      for (int v = 0; v < 8; ++v)
        b.u[v] = *(const uint32_t*)(wT + c * D_ + kbase + v * 2);
      acc[0][nt] = __builtin_amdgcn_wmma_f32_16x16x32_bf16(
          false, a[0].v, false, b.v, (short)0, acc[0][nt], false, false);
      acc[1][nt] = __builtin_amdgcn_wmma_f32_16x16x32_bf16(
          false, a[1].v, false, b.v, (short)0, acc[1][nt], false, false);
    }
  }
#pragma unroll
  for (int mb = 0; mb < 2; ++mb)
#pragma unroll
    for (int nt = 0; nt < 4; ++nt)
#pragma unroll
      for (int r = 0; r < 8; ++r) {
        int m  = m0 + mb * 16 + r + hl * 8;
        int bb = m / S_, s = m % S_;
        int f  = nt * 16 + col;
        uint16_t val = f2bf(acc[mb][nt][r]);
        if (which == 0)      qb[((size_t)(bb * H_ + h) * S_ + s) * F_ + f] = val;
        else if (which == 1) kb[((size_t)(bb * H_ + h) * S_ + s) * F_ + f] = val;
        else                 vT[((size_t)(bb * H_ + h) * F_ + f) * S_ + s] = val; // V transposed
      }
}

// ---------------- row L2-norms (from bf16-rounded values) ----------------
__global__ void k_norm(const uint16_t* __restrict__ m, float* __restrict__ out, int rows) {
  int i = blockIdx.x * blockDim.x + threadIdx.x;
  if (i >= rows) return;
  const uint32_t* p = (const uint32_t*)(m + (size_t)i * F_);
  float s = 0.f;
#pragma unroll
  for (int j = 0; j < 32; ++j) {
    uint32_t w = p[j];
    float a = bf2f((uint16_t)w), b = bf2f((uint16_t)(w >> 16));
    s += a * a + b * b;
  }
  out[i] = s;
}

// ---------------- fused flash L2-attention ----------------
// One wave per 16-query tile per (b,h). Keys streamed in chunks of 32.
// Softmax fully parallel: scores stay in C layout; row max/sum via
// __shfl_xor butterflies (masks 1/2/4/8 stay inside each 16-lane half,
// which holds one row-group of the CDNA5 wave32 C layout).
__global__ void k_attn(const uint16_t* __restrict__ qb, const uint16_t* __restrict__ kb,
                       const uint16_t* __restrict__ vT, const float* __restrict__ qn,
                       const float* __restrict__ kn, uint16_t* __restrict__ cat) {
  const int lane = threadIdx.x;
  const int col  = lane & 15;
  const int hl   = lane >> 4;
  const int m0   = blockIdx.x * 16;
  const int bh   = blockIdx.y;
  const int bb   = bh / H_, h = bh % H_;
  const float rscale = 0.03608439182435161f;   // 1/sqrt(768)

  __shared__ uint16_t sc16[16 * 32];           // probability tile, bf16, [row][key]

  // Q A-operands (K-dim = F = 64 -> two 16x32 chunks), resident for the kernel.
  AB aq[2];
  {
    const uint16_t* qrow = qb + ((size_t)bh * S_ + m0 + col) * F_;
#pragma unroll
    for (int kc2 = 0; kc2 < 2; ++kc2)
#pragma unroll
      for (int v = 0; v < 8; ++v) {
        int kk = kc2 * 32 + hl * 16 + v * 2;
        aq[kc2].u[v] = *(const uint32_t*)(qrow + kk);
      }
  }
  // per-lane copies of ||q||^2 for this half's 8 rows, and running m/l
  float qn8[8], m8[8], l8[8];
#pragma unroll
  for (int r = 0; r < 8; ++r) {
    qn8[r] = qn[(size_t)bh * S_ + m0 + r + hl * 8];
    m8[r]  = -3.0e38f;
    l8[r]  = 0.f;
  }
  v8f O[4] = {};

  for (int t0 = 0; t0 < S_; t0 += 32) {
    float kncol[2];
    kncol[0] = kn[(size_t)bh * S_ + t0 + col];
    kncol[1] = kn[(size_t)bh * S_ + t0 + 16 + col];

    // scores: QK^T for 32 keys -> two 16x16 C tiles, then dist in registers
    v8f d[2];
#pragma unroll
    for (int nt = 0; nt < 2; ++nt) {
      v8f accs = {};
#pragma unroll
      for (int kc2 = 0; kc2 < 2; ++kc2) {
        AB bk;
        const uint16_t* krow = kb + ((size_t)bh * S_ + t0 + nt * 16 + col) * F_;
        int fb = kc2 * 32 + hl * 16;
#pragma unroll
        for (int v = 0; v < 8; ++v)
          bk.u[v] = *(const uint32_t*)(krow + fb + v * 2);
        accs = __builtin_amdgcn_wmma_f32_16x16x32_bf16(
            false, aq[kc2].v, false, bk.v, (short)0, accs, false, false);
      }
#pragma unroll
      for (int r = 0; r < 8; ++r) {
        float sq = qn8[r] + kncol[nt] - 2.0f * accs[r];
        d[nt][r] = sqrtf(fmaxf(sq, 1e-12f)) * rscale;
      }
    }

    // streaming softmax, all 32 lanes busy
#pragma unroll
    for (int r = 0; r < 8; ++r) {
      float rm = fmaxf(d[0][r], d[1][r]);
#pragma unroll
      for (int mk = 1; mk < 16; mk <<= 1)
        rm = fmaxf(rm, __shfl_xor(rm, mk, 32));
      float mnew  = fmaxf(m8[r], rm);
      float scale = __expf(m8[r] - mnew);
      float p0 = __expf(d[0][r] - mnew);
      float p1 = __expf(d[1][r] - mnew);
      sc16[(r + hl * 8) * 32 + col]      = f2bf(p0);
      sc16[(r + hl * 8) * 32 + 16 + col] = f2bf(p1);
      float ps = p0 + p1;
#pragma unroll
      for (int mk = 1; mk < 16; mk <<= 1)
        ps += __shfl_xor(ps, mk, 32);
      l8[r] = l8[r] * scale + ps;
      m8[r] = mnew;
#pragma unroll
      for (int nt = 0; nt < 4; ++nt) O[nt][r] *= scale;   // rescale running output
    }
    __syncthreads();

    // probabilities -> bf16 A-layout (C->A transpose through LDS)
    AB pa;
#pragma unroll
    for (int v = 0; v < 8; ++v) {
      int kk = hl * 16 + v * 2;
      pa.u[v] = *(const uint32_t*)&sc16[col * 32 + kk];
    }
    // P @ V : V transposed so B-operand loads are contiguous dword pairs
#pragma unroll
    for (int nt = 0; nt < 4; ++nt) {
      AB bv;
      const uint16_t* vrow = vT + ((size_t)bh * F_ + nt * 16 + col) * S_;
      int tb = t0 + hl * 16;
#pragma unroll
      for (int v = 0; v < 8; ++v)
        bv.u[v] = *(const uint32_t*)(vrow + tb + v * 2);
      O[nt] = __builtin_amdgcn_wmma_f32_16x16x32_bf16(
          false, pa.v, false, bv.v, (short)0, O[nt], false, false);
    }
    __syncthreads();
  }

  // finalize: divide by row sums (register-resident, consistent per half)
#pragma unroll
  for (int r = 0; r < 8; ++r) {
    float inv = 1.0f / l8[r];
    int s = m0 + r + hl * 8;
#pragma unroll
    for (int nt = 0; nt < 4; ++nt) {
      int f = nt * 16 + col;
      cat[((size_t)(bb * S_ + s)) * OUT_ + h * F_ + f] = f2bf(O[nt][r] * inv);
    }
  }
}

// ---------------- output projection: out = cat @ wo^T + bo ----------------
// One wave computes a 32x64 tile (two M blocks share every B fetch).
__global__ void k_out(const uint16_t* __restrict__ catb, const uint16_t* __restrict__ wob,
                      const float* __restrict__ bo, float* __restrict__ out) {
  const int lane = threadIdx.x;
  const int col  = lane & 15;
  const int hl   = lane >> 4;
  const int m0   = blockIdx.x * 32;
  const int n0   = blockIdx.y * 64;

  v8f acc[2][4] = {};
  for (int kc = 0; kc < OUT_; kc += 32) {
    AB a[2];
#pragma unroll
    for (int mb = 0; mb < 2; ++mb)
#pragma unroll
      for (int v = 0; v < 8; ++v) {
        int kk = kc + hl * 16 + v * 2;
        a[mb].u[v] = *(const uint32_t*)(catb + (size_t)(m0 + mb * 16 + col) * OUT_ + kk);
      }
#pragma unroll
    for (int nt = 0; nt < 4; ++nt) {
      int n = n0 + nt * 16 + col;
      AB bw;
      int kbase = kc + hl * 16;
#pragma unroll
      for (int v = 0; v < 8; ++v)
        bw.u[v] = *(const uint32_t*)(wob + (size_t)n * OUT_ + kbase + v * 2);   // wo[n][k]
      acc[0][nt] = __builtin_amdgcn_wmma_f32_16x16x32_bf16(
          false, a[0].v, false, bw.v, (short)0, acc[0][nt], false, false);
      acc[1][nt] = __builtin_amdgcn_wmma_f32_16x16x32_bf16(
          false, a[1].v, false, bw.v, (short)0, acc[1][nt], false, false);
    }
  }
#pragma unroll
  for (int nt = 0; nt < 4; ++nt) {
    int n = n0 + nt * 16 + col;
    float bias = bo[n];
#pragma unroll
    for (int mb = 0; mb < 2; ++mb)
#pragma unroll
      for (int r = 0; r < 8; ++r) {
        int m = m0 + mb * 16 + r + hl * 8;
        out[(size_t)m * OUT_ + n] = acc[mb][nt][r] + bias;
      }
  }
}

// ---------------- host launcher ----------------
extern "C" void kernel_launch(void* const* d_in, const int* in_sizes, int n_in,
                              void* d_out, int out_size, void* d_ws, size_t ws_size,
                              hipStream_t stream) {
  const float* x  = (const float*)d_in[0];
  const float* wq = (const float*)d_in[1];
  const float* wk = (const float*)d_in[2];
  const float* wv = (const float*)d_in[3];
  const float* wo = (const float*)d_in[4];
  const float* bo = (const float*)d_in[5];
  float* out = (float*)d_out;
  (void)in_sizes; (void)n_in; (void)out_size; (void)ws_size;

  char* ws = (char*)d_ws;
  size_t off = 0;
  auto take = [&](size_t bytes) {
    char* p = ws + off;
    off += (bytes + 255) & ~(size_t)255;
    return p;
  };
  uint16_t* xb   = (uint16_t*)take((size_t)MTOT * D_ * 2);       // x in bf16
  uint16_t* wT   = (uint16_t*)take((size_t)3 * 768 * 768 * 2);   // [2304][768] transposed qkv weights
  uint16_t* wob  = (uint16_t*)take((size_t)768 * 768 * 2);       // wo bf16 (torch [out][in])
  uint16_t* qb   = (uint16_t*)take((size_t)B_ * H_ * S_ * F_ * 2);
  uint16_t* kbb  = (uint16_t*)take((size_t)B_ * H_ * S_ * F_ * 2);
  uint16_t* vT   = (uint16_t*)take((size_t)B_ * H_ * S_ * F_ * 2); // V transposed [b,h,f,s]
  float*    qn   = (float*)take((size_t)B_ * H_ * S_ * 4);
  float*    kn   = (float*)take((size_t)B_ * H_ * S_ * 4);
  uint16_t* catb = (uint16_t*)take((size_t)MTOT * OUT_ * 2);

  k_f32_to_bf16<<<dim3(1024), dim3(256), 0, stream>>>(x, xb, MTOT * D_);
  k_wT_bf16<<<dim3(1024), dim3(256), 0, stream>>>(wq, wT);
  k_wT_bf16<<<dim3(1024), dim3(256), 0, stream>>>(wk, wT + (size_t)768 * 768);
  k_wT_bf16<<<dim3(1024), dim3(256), 0, stream>>>(wv, wT + (size_t)2 * 768 * 768);
  k_f32_to_bf16<<<dim3(1024), dim3(256), 0, stream>>>(wo, wob, 768 * 768);

  k_qkv<<<dim3(MTOT / 32, 36), dim3(32), 0, stream>>>(xb, wT, qb, kbb, vT);

  const int rows = B_ * H_ * S_;
  k_norm<<<dim3((rows + 255) / 256), dim3(256), 0, stream>>>(qb, qn, rows);
  k_norm<<<dim3((rows + 255) / 256), dim3(256), 0, stream>>>(kbb, kn, rows);

  k_attn<<<dim3(S_ / 16, B_ * H_), dim3(32), 0, stream>>>(qb, kbb, vT, qn, kn, catb);

  k_out<<<dim3(MTOT / 32, OUT_ / 64), dim3(32), 0, stream>>>(catb, wob, bo, out);
}
